// TransformerEncoderLayer_4329327034670
// MI455X (gfx1250) — compile-verified, matrix operand order checked by reading, not run on hardware
//
#include <hip/hip_runtime.h>

// ---------------- problem constants ----------------
static constexpr int BB  = 16;    // batch
static constexpr int LLn = 128;   // seq len
static constexpr int DDn = 512;   // model dim
static constexpr int HHn = 8;     // heads
static constexpr int DHn = 64;    // head dim
static constexpr int DFF = 2048;  // ffn dim

typedef __bf16 bf16;
typedef __attribute__((ext_vector_type(16))) __bf16 v16bf;
typedef __attribute__((ext_vector_type(8)))  float  v8f;
typedef __attribute__((ext_vector_type(4)))  int    v4i;
typedef __attribute__((ext_vector_type(8)))  int    i32x8;
typedef __attribute__((ext_vector_type(4)))  int    i32x4;
typedef __attribute__((ext_vector_type(4)))  unsigned u32x4;

// address-space qualified pointee types for the async-LDS builtin
typedef __attribute__((address_space(1))) v4i v4i_g;   // global
typedef __attribute__((address_space(3))) v4i v4i_l;   // LDS

// ---- CDNA5 feature probes (compile-time, with safe fallbacks) ----
#if defined(__has_builtin)
#if __has_builtin(__builtin_amdgcn_global_load_async_to_lds_b128) && \
    __has_builtin(__builtin_amdgcn_s_wait_asynccnt)
#define HAVE_ASYNC_LDS 1
#endif
#if __has_builtin(__builtin_amdgcn_tensor_load_to_lds) && \
    __has_builtin(__builtin_amdgcn_s_wait_tensorcnt)
#define HAVE_TDM 1
#endif
#endif
#ifndef HAVE_ASYNC_LDS
#define HAVE_ASYNC_LDS 0
#endif
#ifndef HAVE_TDM
#define HAVE_TDM 0
#endif

union FragU { v16bf v; unsigned u[8]; };

// ISA 16-bit A-matrix 16x32 layout: lane&15 = row; lanes<16 hold K pairs
// {0,1},{2,3},{4,5},{6,7},{16,17},{18,19},{20,21},{22,23}; lanes>=16 same rows
// with K offset +8.  B fragment mirrored with N in place of M, fed from
// B^T (i.e. [n][k]) row-major storage so pairs are contiguous 32-bit loads.
__device__ __forceinline__ int koff_u32(int v, int lane) {
  return ((lane & 16) ? 4 : 0) + v + ((v >= 4) ? 4 : 0);
}

__device__ __forceinline__ v16bf load_frag(const unsigned* p, int base, int lane) {
  FragU f;
#pragma unroll
  for (int i = 0; i < 8; ++i) f.u[i] = p[base + koff_u32(i, lane)];
  return f.v;
}

__device__ __forceinline__ v16bf load_frag_pred(const unsigned* p, int base, int lane, bool valid) {
  FragU f;
#pragma unroll
  for (int i = 0; i < 8; ++i) f.u[i] = valid ? p[base + koff_u32(i, lane)] : 0u;
  return f.v;
}

__device__ __forceinline__ v8f wmma_bf(v16bf a, v16bf b, v8f c) {
  return __builtin_amdgcn_wmma_f32_16x16x32_bf16(false, a, false, b, (short)0, c, false, false);
}

__device__ __forceinline__ v8f vzero8() {
  v8f z = {0.f,0.f,0.f,0.f,0.f,0.f,0.f,0.f};
  return z;
}

__device__ __forceinline__ unsigned pack2bf(float a, float b) {
  union { bf16 h[2]; unsigned u; } t;
  t.h[0] = (bf16)a; t.h[1] = (bf16)b;
  return t.u;
}

// ---------------- small elementwise kernels ----------------
__global__ __launch_bounds__(256) void cvt_bf16_kernel(const float* __restrict__ s,
                                                       bf16* __restrict__ d, int n) {
  int i = blockIdx.x * 256 + threadIdx.x;
  if (i < n) d[i] = (bf16)s[i];
}

// conv_w [64(o),64(c),4(kk)] -> Wc[o][j = kk*64+c]  (bf16, row-major, K=256)
__global__ __launch_bounds__(256) void pack_convw_kernel(const float* __restrict__ cw,
                                                         bf16* __restrict__ Wc) {
  int i = blockIdx.x * 256 + threadIdx.x;
  if (i >= 64 * 256) return;
  int o = i >> 8, j = i & 255, kk = j >> 6, c = j & 63;
  Wc[i] = (bf16)cw[(o * 64 + c) * 4 + kk];
}

// LayerNorm over rows of 512, fp32 in -> bf16 out.  One wave per row.
__global__ __launch_bounds__(256) void ln512_kernel(const float* __restrict__ x,
                                                    const float* __restrict__ g,
                                                    const float* __restrict__ be,
                                                    bf16* __restrict__ out) {
  int lane = threadIdx.x & 31, wave = threadIdx.x >> 5;
  size_t row = (size_t)blockIdx.x * 8 + wave;
  const float* xr = x + row * DDn;
  float v[16]; float s = 0.f;
#pragma unroll
  for (int i = 0; i < 16; ++i) { v[i] = xr[lane + i * 32]; s += v[i]; }
#pragma unroll
  for (int off = 16; off > 0; off >>= 1) s += __shfl_xor(s, off, 32);
  float mean = s * (1.f / 512.f);
  float s2 = 0.f;
#pragma unroll
  for (int i = 0; i < 16; ++i) { v[i] -= mean; s2 += v[i] * v[i]; }
#pragma unroll
  for (int off = 16; off > 0; off >>= 1) s2 += __shfl_xor(s2, off, 32);
  float rstd = rsqrtf(s2 * (1.f / 512.f) + 1e-6f);
#pragma unroll
  for (int i = 0; i < 16; ++i) {
    int c = lane + i * 32;
    out[row * DDn + c] = (bf16)(v[i] * rstd * g[c] + be[c]);
  }
}

// ---------------- LDS tile staging (async path + sync fallback) -------------
#if HAVE_ASYNC_LDS
__device__ __forceinline__ void stage_tile_async(const bf16* g, int ld, int row0, int kb,
                                                 bf16* lds, int tid) {
  // 128x32 bf16 tile = 8KB: two 16B chunks per thread, DMA'd global->LDS.
#pragma unroll
  for (int i = 0; i < 2; ++i) {
    int o = tid * 16 + i * 4096;          // byte offset in tile
    int row = o >> 6, cb = o & 63;        // 64B per tile row
    const bf16* src = g + (size_t)(row0 + row) * ld + kb + (cb >> 1);
    __builtin_amdgcn_global_load_async_to_lds_b128(
        (v4i_g*)src, (v4i_l*)((char*)lds + o), 0, 0);
  }
}
#endif

__device__ __forceinline__ void stage_tile_sync(const bf16* g, int ld, int row0, int kb,
                                                unsigned* ldsU, int tid) {
#pragma unroll
  for (int i = 0; i < 8; ++i) {
    int p = tid + i * 256;
    int row = p >> 4, c2 = p & 15;
    ldsU[p] = *(const unsigned*)(g + (size_t)(row0 + row) * ld + kb + c2 * 2);
  }
}

// ---------------- generic bf16 WMMA GEMM: C = epilog(A[M,K] * W[N,K]^T) ------
// PERM: 0 -> row-major [M,N]; 1 -> QKV [B,H,L,64]; 2 -> V^T [B,H,64,L]
template <int PERM, bool RELU, bool HASRES, bool OUTF_EN, bool OUTB_EN>
__global__ __launch_bounds__(256)
void gemm_bf16_kernel(const bf16* __restrict__ A, const bf16* __restrict__ W,
                      const float* __restrict__ bias, const float* __restrict__ res,
                      float* __restrict__ outF, bf16* __restrict__ outB,
                      int M, int N, int K, float scale) {
  __shared__ __align__(16) bf16 As[2][128 * 32];
  __shared__ __align__(16) bf16 Bs[2][128 * 32];

  const int tid = threadIdx.x, lane = tid & 31, wave = tid >> 5;
  const int wm = wave >> 1, wn = wave & 1;           // 4x2 wave grid, 32x64 tiles
  const int m0 = blockIdx.y * 128, n0 = blockIdx.x * 128;

  v8f acc[2][4];
#pragma unroll
  for (int i = 0; i < 2; ++i)
#pragma unroll
    for (int j = 0; j < 4; ++j) acc[i][j] = vzero8();

#if HAVE_ASYNC_LDS
  // double-buffered: DMA next K-slice while WMMA consumes current one
  stage_tile_async(A, K, m0, 0, As[0], tid);
  stage_tile_async(W, K, n0, 0, Bs[0], tid);
  int cur = 0;
  for (int kb = 0; kb < K; kb += 32) {
    __builtin_amdgcn_s_wait_asynccnt(0);
    __syncthreads();
    if (kb + 32 < K) {
      stage_tile_async(A, K, m0, kb + 32, As[cur ^ 1], tid);
      stage_tile_async(W, K, n0, kb + 32, Bs[cur ^ 1], tid);
    }
    const unsigned* AsU = (const unsigned*)As[cur];
    const unsigned* BsU = (const unsigned*)Bs[cur];
    v16bf a0 = load_frag(AsU, (wm * 32 +      (lane & 15)) * 16, lane);
    v16bf a1 = load_frag(AsU, (wm * 32 + 16 + (lane & 15)) * 16, lane);
#pragma unroll
    for (int nt = 0; nt < 4; ++nt) {
      v16bf bm = load_frag(BsU, (wn * 64 + nt * 16 + (lane & 15)) * 16, lane);
      acc[0][nt] = wmma_bf(a0, bm, acc[0][nt]);
      acc[1][nt] = wmma_bf(a1, bm, acc[1][nt]);
    }
    cur ^= 1;
  }
#else
  for (int kb = 0; kb < K; kb += 32) {
    stage_tile_sync(A, K, m0, kb, (unsigned*)As[0], tid);
    stage_tile_sync(W, K, n0, kb, (unsigned*)Bs[0], tid);
    __syncthreads();
    const unsigned* AsU = (const unsigned*)As[0];
    const unsigned* BsU = (const unsigned*)Bs[0];
    v16bf a0 = load_frag(AsU, (wm * 32 +      (lane & 15)) * 16, lane);
    v16bf a1 = load_frag(AsU, (wm * 32 + 16 + (lane & 15)) * 16, lane);
#pragma unroll
    for (int nt = 0; nt < 4; ++nt) {
      v16bf bm = load_frag(BsU, (wn * 64 + nt * 16 + (lane & 15)) * 16, lane);
      acc[0][nt] = wmma_bf(a0, bm, acc[0][nt]);
      acc[1][nt] = wmma_bf(a1, bm, acc[1][nt]);
    }
    __syncthreads();
  }
#endif

  const int nl = lane & 15;
  const int mh = (lane & 16) ? 8 : 0;
#pragma unroll
  for (int mt = 0; mt < 2; ++mt)
#pragma unroll
    for (int nt = 0; nt < 4; ++nt) {
      int n = n0 + wn * 64 + nt * 16 + nl;
      float bb = bias[n];
#pragma unroll
      for (int r = 0; r < 8; ++r) {
        int m = m0 + wm * 32 + mt * 16 + r + mh;
        float v = (acc[mt][nt][r] + bb) * scale;
        if constexpr (RELU) v = fmaxf(v, 0.f);
        if constexpr (HASRES) v += res[(size_t)m * N + n];
        if constexpr (OUTF_EN) outF[(size_t)m * N + n] = v;
        if constexpr (OUTB_EN) {
          size_t addr;
          if constexpr (PERM == 0) {
            addr = (size_t)m * N + n;
          } else {
            int b = m >> 7, q = m & 127, h = n >> 6, d = n & 63;
            if constexpr (PERM == 1)
              addr = ((size_t)((b * HHn + h) * LLn + q)) * DHn + d;
            else
              addr = ((size_t)((b * HHn + h) * DHn + d)) * LLn + q;
          }
          outB[addr] = (bf16)v;
        }
      }
    }
}

// ---------------- structure: conv(4x64 contraction) + ReLU + LN(64) ---------
// A row m = (b, e=q*L+k):  A[m][j=kk*64+c] = structure[e,kk,b,c]  (f32->bf16)
// Output: st_kd[b,q,k,d] and st_dk[b,q,d,k] (bf16).
__global__ __launch_bounds__(256)
void struct_kernel(const float* __restrict__ structure, const bf16* __restrict__ Wc,
                   const float* __restrict__ conv_b, const float* __restrict__ g_s,
                   const float* __restrict__ b_s, bf16* __restrict__ st_kd,
                   bf16* __restrict__ st_dk) {
  __shared__ __align__(16) bf16 WcS[64 * 256];
  unsigned* WcU = (unsigned*)WcS;
  const int tid = threadIdx.x, lane = tid & 31, wave = tid >> 5;

#if HAVE_TDM
  // Tensor Data Mover: DMA the whole 64x256 bf16 weight tile (32KB) into LDS
  // with one descriptor.  D# built per CDNA5 ISA ch.8 (2D, data_size=2B).
  if (wave == 0) {
    unsigned ldsoff =
        (unsigned)(size_t)(__attribute__((address_space(3))) char*)WcS;
    unsigned long long ga = (unsigned long long)(size_t)Wc;
    u32x4 g0 = {1u,                                   // count=1 (user D#)
                ldsoff,                               // lds_addr
                (unsigned)(ga & 0xFFFFFFFFu),         // global_addr[31:0]
                (unsigned)((ga >> 32) & 0x01FFFFFFu) | (2u << 30)}; // [56:32]|type=2
    i32x8 g1 = {(int)(1u << 16),          // data_size=2B; wg_mask=0; no pad/iter
                (int)(16384u << 16),      // tensor_dim0 = 16384 elements
                (int)(1u << 16),          // tensor_dim1 = 1
                (int)(16384u << 16),      // tile_dim0 = 16384
                1,                        // tile_dim1 = 1, tile_dim2 = 0
                16384,                    // tensor_dim0_stride
                0, 0};                    // tensor_dim1_stride = 0
    i32x4 gz4 = {0, 0, 0, 0};
    i32x8 gz8 = {0, 0, 0, 0, 0, 0, 0, 0};
    __builtin_amdgcn_tensor_load_to_lds(g0, g1, gz4, gz4, gz8, 0);
    __builtin_amdgcn_s_wait_tensorcnt(0);
  }
#else
#pragma unroll
  for (int i = 0; i < 32; ++i) WcU[tid + i * 256] = ((const unsigned*)Wc)[tid + i * 256];
#endif
  __syncthreads();

  const int m0 = (blockIdx.x * 8 + wave) * 16;
  const int row_m = m0 + (lane & 15);
  const int bI = row_m >> 14;            // / (L*L)
  const int e  = row_m & 16383;
  const int kbase = (lane & 16) ? 8 : 0;

  // prefetch this lane's four 256B structure rows (streams 256MB exactly once)
#pragma unroll
  for (int kk = 0; kk < 4; ++kk)
    __builtin_prefetch(structure + (((size_t)e * 4 + kk) * BB + bI) * DHn, 0, 1);

  v8f acc[4];
#pragma unroll
  for (int i = 0; i < 4; ++i) acc[i] = vzero8();

  for (int ks = 0; ks < 8; ++ks) {       // K = 256
    FragU a;
#pragma unroll
    for (int i = 0; i < 8; ++i) {
      int j = ks * 32 + kbase + 2 * i + ((i >= 4) ? 8 : 0);
      int kk = j >> 6, c = j & 63;
      const float* p = structure + (((size_t)e * 4 + kk) * BB + bI) * DHn + c;
      a.u[i] = pack2bf(p[0], p[1]);
    }
#pragma unroll
    for (int nt = 0; nt < 4; ++nt) {     // Wc row length = 128 u32
      v16bf bm = load_frag(WcU, (nt * 16 + (lane & 15)) * 128 + ks * 16, lane);
      acc[nt] = wmma_bf(a.v, bm, acc[nt]);
    }
  }

  const int nl = lane & 15;
  const int mh = (lane & 16) ? 8 : 0;
  float cb[4], gg[4], bbv[4];
#pragma unroll
  for (int nt = 0; nt < 4; ++nt) {
    int n = nt * 16 + nl;
    cb[nt] = conv_b[n]; gg[nt] = g_s[n]; bbv[nt] = b_s[n];
  }
#pragma unroll
  for (int r = 0; r < 8; ++r) {
    int m = m0 + r + mh;
    float v[4]; float s1 = 0.f, s2 = 0.f;
#pragma unroll
    for (int nt = 0; nt < 4; ++nt) {
      v[nt] = fmaxf(acc[nt][r] + cb[nt], 0.f);   // ReLU before LN (as reference)
      s1 += v[nt]; s2 += v[nt] * v[nt];
    }
#pragma unroll
    for (int off = 1; off < 16; off <<= 1) {     // reduce within 16-lane half
      s1 += __shfl_xor(s1, off, 32);
      s2 += __shfl_xor(s2, off, 32);
    }
    float mean = s1 * (1.f / 64.f);
    float var  = s2 * (1.f / 64.f) - mean * mean;
    float rstd = rsqrtf(var + 1e-6f);
    int b2 = m >> 14, q = (m >> 7) & 127, k = m & 127;
#pragma unroll
    for (int nt = 0; nt < 4; ++nt) {
      int n = nt * 16 + nl;
      float o = (v[nt] - mean) * rstd * gg[nt] + bbv[nt];
      st_kd[(size_t)m * DHn + n] = (bf16)o;
      st_dk[((size_t)(b2 * LLn + q) * DHn + n) * LLn + k] = (bf16)o;
    }
  }
}

// ---------------- fused relation-aware attention ----------------------------
// Block = (q-tile of 16, batch b, head-group hg of 4 heads). 8 waves.
// scores = QK^T + Q.st_kd -> mask -> softmax -> ctx = A.V + A.st_dk, all in LDS.
__global__ __launch_bounds__(256)
void attn_kernel(const bf16* __restrict__ q_bf, const bf16* __restrict__ k_bf,
                 const bf16* __restrict__ vT_bf, const bf16* __restrict__ st_kd,
                 const bf16* __restrict__ st_dk, const int* __restrict__ mask,
                 bf16* __restrict__ ctxb) {
  __shared__ __align__(16) char smem[57344];
  bf16*  qs   = (bf16*)smem;               // [4][16][64]   (phases 0..1b)
  float* S    = (float*)(smem + 8192);     // [4][16][128]  (phases 1..2)
  bf16*  Aatt = (bf16*)(smem + 40960);     // [4][16][128]  (phases 2..3b)
  float* Cs   = (float*)(smem + 8192);     // [16][4][64]   (aliases S; S dead)
  unsigned* qsU = (unsigned*)qs;

  const int b = blockIdx.y, q0 = blockIdx.x * 16, hg = blockIdx.z;
  const int tid = threadIdx.x, lane = tid & 31, wave = tid >> 5;

  // phase 0: stage Q tile for 4 heads (bf16, already scaled by 1/8)
#pragma unroll
  for (int i = 0; i < 8; ++i) {
    int p = tid + i * 256;                 // 2048 u32
    int hl = p >> 9, rem = p & 511, ql = rem >> 5, c2 = rem & 31;
    qsU[p] = *(const unsigned*)(q_bf +
        (((size_t)(b * HHn + hg * 4 + hl) * LLn) + q0 + ql) * DHn + 2 * c2);
  }
  __syncthreads();

  // phase 1: QK^T.  wave = (head, half of k-tiles)
  {
    int hl = wave & 3, h = hg * 4 + hl;
    int kts = (wave >> 2) * 4;
    const unsigned* qrow = qsU + (hl * 16 + (lane & 15)) * 32;
    for (int kt = kts; kt < kts + 4; ++kt) {
      v8f acc = vzero8();
      const unsigned* krow = (const unsigned*)(k_bf +
          ((size_t)(b * HHn + h) * LLn + kt * 16 + (lane & 15)) * DHn);
#pragma unroll
      for (int dss = 0; dss < 2; ++dss) {
        v16bf a = load_frag(qrow, dss * 16, lane);
        v16bf bm = load_frag(krow, dss * 16, lane);
        acc = wmma_bf(a, bm, acc);
      }
      int mh = (lane & 16) ? 8 : 0, nl = lane & 15;
#pragma unroll
      for (int r = 0; r < 8; ++r)
        S[(hl * 16 + r + mh) * 128 + kt * 16 + nl] = acc[r];
    }
  }
  __syncthreads();

  // phase 1b: relation score bias.  Per q: [heads(4 pad 16) x 64] * st_kd^T.
  for (int t = 0; t < 2; ++t) {
    int ql = wave * 2 + t, q = q0 + ql;
    int mrow = lane & 15;
    bool avalid = mrow < 4;
    const unsigned* arow = qsU + (mrow * 16 + ql) * 32;
    for (int kt = 0; kt < 8; ++kt) {
      v8f acc = vzero8();
      const unsigned* brow = (const unsigned*)(st_kd +
          (((size_t)b * LLn + q) * LLn + kt * 16 + (lane & 15)) * DHn);
#pragma unroll
      for (int dss = 0; dss < 2; ++dss) {
        v16bf a = load_frag_pred(arow, dss * 16, lane, avalid);
        v16bf bm = load_frag(brow, dss * 16, lane);
        acc = wmma_bf(a, bm, acc);
      }
      if (!(lane & 16)) {                  // rows 0..3 = valid heads
        int nl = lane & 15;
#pragma unroll
        for (int r = 0; r < 4; ++r)
          S[(r * 16 + ql) * 128 + kt * 16 + nl] += acc[r];
      }
    }
  }
  __syncthreads();

  // phase 2: mask + softmax over k (128), write bf16 attn
#pragma unroll 1
  for (int j = 0; j < 8; ++j) {
    int row = wave * 8 + j;                // 64 rows total
    int ql = row & 15, q = q0 + ql;
    float x[4];
#pragma unroll
    for (int i = 0; i < 4; ++i) {
      int k = lane + 32 * i;
      float sv = S[row * 128 + k];
      if (mask[((size_t)b * LLn + q) * LLn + k] != 0) sv = -1e18f;
      x[i] = sv;
    }
    float mx = fmaxf(fmaxf(x[0], x[1]), fmaxf(x[2], x[3]));
#pragma unroll
    for (int off = 16; off > 0; off >>= 1) mx = fmaxf(mx, __shfl_xor(mx, off, 32));
    float s = 0.f;
#pragma unroll
    for (int i = 0; i < 4; ++i) { x[i] = __expf(x[i] - mx); s += x[i]; }
#pragma unroll
    for (int off = 16; off > 0; off >>= 1) s += __shfl_xor(s, off, 32);
    float inv = 1.f / s;
#pragma unroll
    for (int i = 0; i < 4; ++i)
      Aatt[row * 128 + lane + 32 * i] = (bf16)(x[i] * inv);
  }
  __syncthreads();

  // phase 3: ctx = attn @ V.  wave = (head, half of d-tiles)
  {
    int hl = wave & 3, h = hg * 4 + hl;
    int dts = (wave >> 2) * 2;
    const unsigned* arow = (const unsigned*)Aatt + (hl * 16 + (lane & 15)) * 64;
    for (int dt = dts; dt < dts + 2; ++dt) {
      v8f acc = vzero8();
      const unsigned* brow = (const unsigned*)(vT_bf +
          ((size_t)(b * HHn + h) * DHn + dt * 16 + (lane & 15)) * LLn);
#pragma unroll
      for (int ks = 0; ks < 4; ++ks) {
        v16bf a = load_frag(arow, ks * 16, lane);
        v16bf bm = load_frag(brow, ks * 16, lane);
        acc = wmma_bf(a, bm, acc);
      }
      int mh = (lane & 16) ? 8 : 0, nl = lane & 15;
#pragma unroll
      for (int r = 0; r < 8; ++r)
        Cs[((r + mh) * 4 + hl) * 64 + dt * 16 + nl] = acc[r];
    }
  }
  __syncthreads();

  // phase 3b: ctx += attn @ st_dk (per q, heads padded to 16)
  for (int t = 0; t < 2; ++t) {
    int ql = wave * 2 + t, q = q0 + ql;
    int mrow = lane & 15;
    bool avalid = mrow < 4;
    const unsigned* arow = (const unsigned*)Aatt + (mrow * 16 + ql) * 64;
    for (int dt = 0; dt < 4; ++dt) {
      v8f acc = vzero8();
      const unsigned* brow = (const unsigned*)(st_dk +
          (((size_t)b * LLn + q) * DHn + dt * 16 + (lane & 15)) * LLn);
#pragma unroll
      for (int ks = 0; ks < 4; ++ks) {
        v16bf a = load_frag_pred(arow, ks * 16, lane, avalid);
        v16bf bm = load_frag(brow, ks * 16, lane);
        acc = wmma_bf(a, bm, acc);
      }
      if (!(lane & 16)) {
        int nl = lane & 15;
#pragma unroll
        for (int r = 0; r < 4; ++r)
          Cs[(ql * 4 + r) * 64 + dt * 16 + nl] += acc[r];
      }
    }
  }
  __syncthreads();

  // phase 4: write ctx as bf16 into [B, L, h*64+d]
  unsigned* ctxU = (unsigned*)ctxb;
#pragma unroll
  for (int i = 0; i < 8; ++i) {
    int p = tid + i * 256;                 // 2048 u32
    int ql = p >> 7, c2 = p & 127;
    int c = c2 * 2, hl = c >> 6, d = c & 63;
    float v0 = Cs[(ql * 4 + hl) * 64 + d];
    float v1 = Cs[(ql * 4 + hl) * 64 + d + 1];
    ctxU[((size_t)(b * LLn) + q0 + ql) * 256 + hg * 128 + c2] = pack2bf(v0, v1);
  }
}

// ---------------- host-side orchestration ----------------------------------
extern "C" void kernel_launch(void* const* d_in, const int* in_sizes, int n_in,
                              void* d_out, int out_size, void* d_ws, size_t ws_size,
                              hipStream_t stream) {
  (void)in_sizes; (void)n_in; (void)out_size; (void)ws_size;
  const float* inputs    = (const float*)d_in[0];
  const float* structure = (const float*)d_in[1];
  const int*   maskp     = (const int*)d_in[2];
  const float* Wq = (const float*)d_in[3];   const float* bq = (const float*)d_in[4];
  const float* Wk = (const float*)d_in[5];   const float* bk = (const float*)d_in[6];
  const float* Wv = (const float*)d_in[7];   const float* bv = (const float*)d_in[8];
  const float* Wo = (const float*)d_in[9];   const float* bo = (const float*)d_in[10];
  const float* conv_w = (const float*)d_in[11]; const float* conv_b = (const float*)d_in[12];
  const float* g_att = (const float*)d_in[13];  const float* b_att = (const float*)d_in[14];
  const float* g_ffn = (const float*)d_in[15];  const float* b_ffn = (const float*)d_in[16];
  const float* g_s   = (const float*)d_in[17];  const float* b_s   = (const float*)d_in[18];
  const float* W1 = (const float*)d_in[19];  const float* b1 = (const float*)d_in[20];
  const float* W2 = (const float*)d_in[21];  const float* b2 = (const float*)d_in[22];

  char* ws = (char*)d_ws; size_t off = 0;
  auto alloc = [&](size_t bytes) -> void* {
    void* p = ws + off; off += (bytes + 255) & ~(size_t)255; return p;
  };
  const int M = BB * LLn;                          // 2048 token rows
  bf16* Wq_b = (bf16*)alloc((size_t)DDn * DDn * 2);
  bf16* Wk_b = (bf16*)alloc((size_t)DDn * DDn * 2);
  bf16* Wv_b = (bf16*)alloc((size_t)DDn * DDn * 2);
  bf16* Wo_b = (bf16*)alloc((size_t)DDn * DDn * 2);
  bf16* W1_b = (bf16*)alloc((size_t)DFF * DDn * 2);
  bf16* W2_b = (bf16*)alloc((size_t)DDn * DFF * 2);
  bf16* Wc_b = (bf16*)alloc((size_t)64 * 256 * 2);
  bf16* xln_b = (bf16*)alloc((size_t)M * DDn * 2);
  bf16* q_b   = (bf16*)alloc((size_t)M * DDn * 2);     // [B,H,L,64]
  bf16* k_b   = (bf16*)alloc((size_t)M * DDn * 2);     // [B,H,L,64]
  bf16* vT_b  = (bf16*)alloc((size_t)M * DDn * 2);     // [B,H,64,L]
  bf16* st_kd = (bf16*)alloc((size_t)BB * LLn * LLn * DHn * 2);
  bf16* st_dk = (bf16*)alloc((size_t)BB * LLn * LLn * DHn * 2);
  bf16* ctx_b = (bf16*)alloc((size_t)M * DDn * 2);
  float* hbuf = (float*)alloc((size_t)M * DDn * 4);
  bf16* yln_b = (bf16*)alloc((size_t)M * DDn * 2);
  bf16* f1_b  = (bf16*)alloc((size_t)M * DFF * 2);

  const int T = 256;
  // weight packing
  cvt_bf16_kernel<<<(DDn * DDn) / T, T, 0, stream>>>(Wq, Wq_b, DDn * DDn);
  cvt_bf16_kernel<<<(DDn * DDn) / T, T, 0, stream>>>(Wk, Wk_b, DDn * DDn);
  cvt_bf16_kernel<<<(DDn * DDn) / T, T, 0, stream>>>(Wv, Wv_b, DDn * DDn);
  cvt_bf16_kernel<<<(DDn * DDn) / T, T, 0, stream>>>(Wo, Wo_b, DDn * DDn);
  cvt_bf16_kernel<<<(DFF * DDn) / T, T, 0, stream>>>(W1, W1_b, DFF * DDn);
  cvt_bf16_kernel<<<(DFF * DDn) / T, T, 0, stream>>>(W2, W2_b, DDn * DFF);
  pack_convw_kernel<<<(64 * 256) / T, T, 0, stream>>>(conv_w, Wc_b);

  // pre-attention LN
  ln512_kernel<<<M / 8, T, 0, stream>>>(inputs, g_att, b_att, xln_b);

  // structure conv + relu + LN -> both layouts (streams 256MB of fp32 once)
  struct_kernel<<<(BB * LLn * LLn) / 128, T, 0, stream>>>(structure, Wc_b, conv_b,
                                                          g_s, b_s, st_kd, st_dk);

  // Q/K/V projections (Q pre-scaled by 1/sqrt(64))
  gemm_bf16_kernel<1, false, false, false, true>
      <<<dim3(4, 16), T, 0, stream>>>(xln_b, Wq_b, bq, nullptr, nullptr, q_b,
                                      M, DDn, DDn, 0.125f);
  gemm_bf16_kernel<1, false, false, false, true>
      <<<dim3(4, 16), T, 0, stream>>>(xln_b, Wk_b, bk, nullptr, nullptr, k_b,
                                      M, DDn, DDn, 1.f);
  gemm_bf16_kernel<2, false, false, false, true>
      <<<dim3(4, 16), T, 0, stream>>>(xln_b, Wv_b, bv, nullptr, nullptr, vT_b,
                                      M, DDn, DDn, 1.f);

  // fused relation-aware attention
  attn_kernel<<<dim3(LLn / 16, BB, 2), T, 0, stream>>>(q_b, k_b, vT_b, st_kd, st_dk,
                                                       maskp, ctx_b);

  // output projection + residual (h = attn_out + inputs, fp32)
  gemm_bf16_kernel<0, false, true, true, false>
      <<<dim3(4, 16), T, 0, stream>>>(ctx_b, Wo_b, bo, inputs, hbuf, nullptr,
                                      M, DDn, DDn, 1.f);
  // FFN
  ln512_kernel<<<M / 8, T, 0, stream>>>(hbuf, g_ffn, b_ffn, yln_b);
  gemm_bf16_kernel<0, true, false, false, true>
      <<<dim3(16, 16), T, 0, stream>>>(yln_b, W1_b, b1, nullptr, nullptr, f1_b,
                                       M, DFF, DDn, 1.f);
  gemm_bf16_kernel<0, false, true, true, false>
      <<<dim3(4, 16), T, 0, stream>>>(f1_b, W2_b, b2, hbuf, (float*)d_out,
                                      nullptr, M, DDn, DFF, 1.f);
}
